// EncoderDecoderModel_51419348468233
// MI455X (gfx1250) — compile-verified, matrix operand order checked by reading, not run on hardware
//
#include <hip/hip_runtime.h>
#include <hip/hip_bf16.h>

typedef _Float16 h16;
typedef __attribute__((ext_vector_type(16))) _Float16 v16h;
typedef __attribute__((ext_vector_type(8)))  float    v8f;
typedef __attribute__((ext_vector_type(4)))  unsigned int u32x4;
typedef __attribute__((ext_vector_type(8)))  int i32x8;
typedef __attribute__((ext_vector_type(4)))  int i32x4;

#if __has_builtin(__builtin_amdgcn_tensor_load_to_lds) && __has_builtin(__builtin_amdgcn_s_wait_tensorcnt)
#define HAS_TDM 1
#else
#define HAS_TDM 0
#endif

// ---------------- model dims ----------------
#define CC     256
#define S_C1H  62
#define S_C1W  511
#define S_P1H  31
#define S_P1W  255
#define S_C2H  29
#define S_C2W  253
#define S_C2M  (S_C2H * S_C2W)      // 7337
#define S_P2H  14
#define S_P2W  126
#define FLATD  (S_P2W * CC)         // 32256
#define TS     14
#define W_C1   254
#define W_P1   127
#define W_C2   125
#define W_P2   62
#define VV     128
#define LL     300
#define EU     256
#define DU     256

__device__ __forceinline__ float sigf(float x) { return 1.f / (1.f + __expf(-x)); }

#if HAS_TDM
// ---- Tensor Data Mover: 2D f16 tile load Global->LDS with row padding ----
// D# layout per CDNA5 ISA.
__device__ __forceinline__ void tdm_load_2d_f16(
    unsigned lds_off, const void* gaddr,
    unsigned tileD0 /*elems per row*/, unsigned tileD1 /*rows*/,
    unsigned tdim0 /*tensor row len*/, unsigned tdim1 /*tensor rows*/,
    unsigned stride0 /*elems*/,
    unsigned padInterval /*code*/, unsigned padAmount /*code*/, int padEnable) {
    unsigned long long ga = (unsigned long long)gaddr;
    u32x4 g0;
    g0[0] = 1u;                                               // count=1, user descriptor
    g0[1] = lds_off;                                          // lds_addr
    g0[2] = (unsigned)(ga & 0xFFFFFFFFu);                     // global_addr[31:0]
    g0[3] = (unsigned)((ga >> 32) & 0x1FFFFFFu) | (2u << 30); // global_addr[56:32] | type=2
    i32x8 g1;
    g1[0] = (int)((1u << 16) |                                // data_size = 1 -> 2 bytes
                  (padEnable ? (1u << 20) : 0u) |
                  ((padInterval & 7u) << 22) |
                  ((padAmount & 127u) << 25));
    g1[1] = (int)((tdim0 & 0xFFFFu) << 16);                   // tensor_dim0[15:0]
    g1[2] = (int)(((tdim0 >> 16) & 0xFFFFu) | ((tdim1 & 0xFFFFu) << 16));
    g1[3] = (int)(((tdim1 >> 16) & 0xFFFFu) | ((tileD0 & 0xFFFFu) << 16)); // tile_dim0
    g1[4] = (int)(tileD1 & 0xFFFFu);                          // tile_dim1 (tile_dim2 = 0)
    g1[5] = (int)stride0;                                     // tensor_dim0_stride[31:0]
    g1[6] = 0;
    g1[7] = 0;
    i32x4 gz = {0, 0, 0, 0};
#if __clang_major__ >= 23
    i32x8 gz8 = {0, 0, 0, 0, 0, 0, 0, 0};
    __builtin_amdgcn_tensor_load_to_lds(g0, g1, gz, gz, gz8, 0);
#else
    __builtin_amdgcn_tensor_load_to_lds(g0, g1, gz, gz, 0);
#endif
}
#endif

// ---------------- utility kernels ----------------
__global__ void k_zero_f32(float* p, int n) {
    int i = blockIdx.x * blockDim.x + threadIdx.x;
    if (i < n) p[i] = 0.f;
}

__global__ void k_f32_to_f16(const float* __restrict__ s, h16* __restrict__ d, int n) {
    int i = blockIdx.x * blockDim.x + threadIdx.x;
    if (i < n) d[i] = (h16)s[i];
}

// s[R][C] -> d[C][R] (f16): weight panels are pre-transposed ONCE so GEMM tiles are
// plain row-major 2D tiles on both sides (TDM-friendly, no in-loop transpose).
__global__ void k_transpose_f32_to_f16(const float* __restrict__ s, h16* __restrict__ d,
                                       int R, int C) {
    int i = blockIdx.x * blockDim.x + threadIdx.x;
    if (i >= R * C) return;
    int r = i / C, c = i % C;
    d[c * R + r] = (h16)s[i];
}

// ---------------- STFT conv1 (1 input channel, 3x3, VALID, relu) ----------------
__global__ void k_conv2d_s1(const float* __restrict__ x, const float* __restrict__ w,
                            const float* __restrict__ b, float* __restrict__ y) {
    int idx = blockIdx.x * blockDim.x + threadIdx.x;
    if (idx >= S_C1H * S_C1W * CC) return;
    int oc = idx & 255;
    int ox = (idx >> 8) % S_C1W;
    int oy = idx / (S_C1W * CC);
    float acc = b[oc];
#pragma unroll
    for (int ky = 0; ky < 3; ++ky)
#pragma unroll
        for (int kx = 0; kx < 3; ++kx)
            acc += x[(oy + ky) * 513 + ox + kx] * w[(ky * 3 + kx) * CC + oc];
    y[idx] = fmaxf(acc, 0.f);
}

// ---------------- 2x2 maxpool (optional f32 / f16 outputs) ----------------
__global__ void k_maxpool2d(const float* __restrict__ x, float* __restrict__ yf,
                            h16* __restrict__ yh, int H, int W, int C) {
    int OH = H >> 1, OW = W >> 1;
    int idx = blockIdx.x * blockDim.x + threadIdx.x;
    if (idx >= OH * OW * C) return;
    int c = idx % C;
    int ox = (idx / C) % OW;
    int oy = idx / (C * OW);
    const float* p = x + ((size_t)(oy * 2) * W + ox * 2) * C + c;
    float m = fmaxf(fmaxf(p[0], p[C]), fmaxf(p[(size_t)W * C], p[(size_t)W * C + C]));
    if (yf) yf[idx] = m;
    if (yh) yh[idx] = (h16)m;
}

__global__ void k_maxpool1d(const float* __restrict__ x, float* __restrict__ yf,
                            h16* __restrict__ yh, int T, int C) {
    int OT = T >> 1;
    int idx = blockIdx.x * blockDim.x + threadIdx.x;
    if (idx >= OT * C) return;
    int c = idx % C, ot = idx / C;
    float m = fmaxf(x[(size_t)(ot * 2) * C + c], x[(size_t)(ot * 2 + 1) * C + c]);
    if (yf) yf[idx] = m;
    if (yh) yh[idx] = (h16)m;
}

// ---------------- im2col (3x3 2D over f16 feature map) ----------------
__global__ void k_im2col2d_f16(const h16* __restrict__ src, h16* __restrict__ dst,
                               int SW, int C, int OH, int OW) {
    size_t total = (size_t)OH * OW * 9 * C;
    size_t idx = (size_t)blockIdx.x * blockDim.x + threadIdx.x;
    if (idx >= total) return;
    int KC = 9 * C;
    int q = (int)(idx % KC);
    int p = (int)(idx / KC);
    int ic = q % C, kk = q / C, ky = kk / 3, kx = kk % 3;
    int oy = p / OW, ox = p % OW;
    dst[idx] = src[((size_t)(oy + ky) * SW + (ox + kx)) * C + ic];
}

__global__ void k_im2col1d_f32(const float* __restrict__ src, h16* __restrict__ dst,
                               int SC, int OT) {
    size_t total = (size_t)OT * 3 * SC;
    size_t idx = (size_t)blockIdx.x * blockDim.x + threadIdx.x;
    if (idx >= total) return;
    int q = (int)(idx % (3 * SC));
    int p = (int)(idx / (3 * SC));
    int ic = q % SC, kw = q / SC;
    dst[idx] = (h16)src[(size_t)(p + kw) * SC + ic];
}

__global__ void k_im2col1d_f16(const h16* __restrict__ src, h16* __restrict__ dst,
                               int SC, int OT) {
    size_t total = (size_t)OT * 3 * SC;
    size_t idx = (size_t)blockIdx.x * blockDim.x + threadIdx.x;
    if (idx >= total) return;
    int q = (int)(idx % (3 * SC));
    int p = (int)(idx / (3 * SC));
    int ic = q % SC, kw = q / SC;
    dst[idx] = src[(size_t)(p + kw) * SC + ic];
}

// ---------------- generic WMMA f16 GEMM: C[M,N] = act(A[M,K] @ Bt^T[K,N] + bias) ----------------
// A: [M,K] row-major f16.  Bt: [N,K] row-major f16 (weights pre-transposed once).
// WG = 256 threads = 8 waves; WG tile 32x128; wave tile 16x32 (two 16x16 accumulators).
// Both tiles are fetched by the Tensor Data Mover with pad_interval=64B/pad_amount=16B so
// rows land at stride 40 halves in LDS; double-buffered so tile i+1 DMA overlaps compute i.
// Requires: K % 32 == 0, N % 128 == 0 (true for every call site).
__global__ __launch_bounds__(256) void k_gemm_f16_wmma(
    const h16* __restrict__ A, const h16* __restrict__ Bt,
    const float* __restrict__ bias, float* __restrict__ C,
    int M, int N, int K, int relu) {
    __shared__ h16 sA[2][32 * 40];
    __shared__ h16 sB[2][128 * 40];
    int tid = threadIdx.x;
    int wave = tid >> 5, lane = tid & 31;
    int m0 = blockIdx.y * 32;
    int n0 = blockIdx.x * 128;
    int wm = (wave >> 2) * 16;      // 0 / 16
    int wn = (wave & 3) * 32;       // 0,32,64,96
    int lr = lane & 15;
    int kh = (lane < 16) ? 0 : 16;
    v8f acc0 = {};
    v8f acc1 = {};
    int nk = K >> 5;
#if HAS_TDM
    if (wave == 0) {
        tdm_load_2d_f16((unsigned)(unsigned long long)(void*)&sA[0][0],
                        A + (size_t)m0 * K, 32, 32,
                        (unsigned)K, (unsigned)(M - m0), (unsigned)K, 3, 3, 1);
        tdm_load_2d_f16((unsigned)(unsigned long long)(void*)&sB[0][0],
                        Bt + (size_t)n0 * K, 32, 128,
                        (unsigned)K, (unsigned)(N - n0), (unsigned)K, 3, 3, 1);
    }
#else
    {
#pragma unroll
        for (int i = tid; i < 32 * 32; i += 256) {
            int r = i >> 5, c = i & 31;
            int gm = m0 + r;
            int gmc = gm < M ? gm : M - 1;
            h16 v = A[(size_t)gmc * K + c];
            sA[0][r * 40 + c] = (gm < M) ? v : (h16)0.0f;
        }
#pragma unroll
        for (int i = tid; i < 128 * 32; i += 256) {
            int r = i >> 5, c = i & 31;
            sB[0][r * 40 + c] = Bt[(size_t)(n0 + r) * K + c];
        }
    }
#endif
    for (int i = 0; i < nk; ++i) {
        int p = i & 1;
        int kk = (i + 1) << 5;      // next slab's K offset
#if HAS_TDM
        if (wave == 0) __builtin_amdgcn_s_wait_tensorcnt(0);  // tile i landed
#endif
        __syncthreads();            // tile i visible; everyone done reading buffer p^1
        if (i + 1 < nk) {
#if HAS_TDM
            if (wave == 0) {        // DMA tile i+1 while computing tile i
                tdm_load_2d_f16((unsigned)(unsigned long long)(void*)&sA[p ^ 1][0],
                                A + (size_t)m0 * K + kk, 32, 32,
                                (unsigned)(K - kk), (unsigned)(M - m0), (unsigned)K, 3, 3, 1);
                tdm_load_2d_f16((unsigned)(unsigned long long)(void*)&sB[p ^ 1][0],
                                Bt + (size_t)n0 * K + kk, 32, 128,
                                (unsigned)(K - kk), (unsigned)(N - n0), (unsigned)K, 3, 3, 1);
            }
#else
#pragma unroll
            for (int j = tid; j < 32 * 32; j += 256) {
                int r = j >> 5, c = j & 31;
                int gm = m0 + r;
                int gmc = gm < M ? gm : M - 1;
                h16 v = A[(size_t)gmc * K + kk + c];
                sA[p ^ 1][r * 40 + c] = (gm < M) ? v : (h16)0.0f;
            }
#pragma unroll
            for (int j = tid; j < 128 * 32; j += 256) {
                int r = j >> 5, c = j & 31;
                sB[p ^ 1][r * 40 + c] = Bt[(size_t)(n0 + r) * K + kk + c];
            }
#endif
        }
        v16h a, b0, b1;
#pragma unroll
        for (int q = 0; q < 16; ++q) a[q]  = sA[p][(wm + lr) * 40 + kh + q];
#pragma unroll
        for (int q = 0; q < 16; ++q) b0[q] = sB[p][(wn + lr) * 40 + kh + q];
#pragma unroll
        for (int q = 0; q < 16; ++q) b1[q] = sB[p][(wn + 16 + lr) * 40 + kh + q];
        acc0 = __builtin_amdgcn_wmma_f32_16x16x32_f16(false, a, false, b0, (short)0, acc0, false, false);
        acc1 = __builtin_amdgcn_wmma_f32_16x16x32_f16(false, a, false, b1, (short)0, acc1, false, false);
    }
    int mb = (lane < 16) ? 0 : 8;
#pragma unroll
    for (int r = 0; r < 8; ++r) {
        int gm = m0 + wm + mb + r;
        if (gm >= M) continue;
        int gn0 = n0 + wn + lr;
        float v0 = acc0[r] + (bias ? bias[gn0] : 0.f);
        C[(size_t)gm * N + gn0] = relu ? fmaxf(v0, 0.f) : v0;
        int gn1 = n0 + wn + 16 + lr;
        float v1 = acc1[r] + (bias ? bias[gn1] : 0.f);
        C[(size_t)gm * N + gn1] = relu ? fmaxf(v1, 0.f) : v1;
    }
}

// ---------------- skinny (M=14) f32 projection: zx = x @ s_Wi + s_b ----------------
// Pure bandwidth: streams the 132MB s_Wi exactly once (~5.7us at 23.3TB/s).
__global__ __launch_bounds__(256) void k_stft_zpre(
    const float* __restrict__ x,   // [14, 32256]
    const float* __restrict__ Wi,  // [32256, 1024]
    const float* __restrict__ b,   // [1024]
    float* __restrict__ zx) {      // [14, 1024]
    int j = blockIdx.x * 256 + threadIdx.x;
    float acc[TS];
#pragma unroll
    for (int t = 0; t < TS; ++t) acc[t] = 0.f;
    for (int k = 0; k < FLATD; ++k) {
        float w = Wi[(size_t)k * 1024 + j];
#pragma unroll
        for (int t = 0; t < TS; ++t) acc[t] += x[(size_t)t * FLATD + k] * w;  // uniform -> SMEM
    }
#pragma unroll
    for (int t = 0; t < TS; ++t) zx[t * 1024 + j] = acc[t] + b[j];
}

// ---------------- sequential LSTM (encoders): zx precomputed; returns final (h,c) ----------------
__global__ __launch_bounds__(1024) void k_lstm_seq(
    const float* __restrict__ zx,  // [T, 1024]
    const float* __restrict__ Wh,  // [256, 1024]
    float* __restrict__ hT, float* __restrict__ cT, int T) {
    __shared__ float sh[EU], sc[EU], sz[1024];
    int tid = threadIdx.x;
    if (tid < EU) { sh[tid] = 0.f; sc[tid] = 0.f; }
    __syncthreads();
    for (int t = 0; t < T; ++t) {
        float acc = zx[(size_t)t * 1024 + tid];
#pragma unroll 4
        for (int k = 0; k < EU; ++k) acc += sh[k] * Wh[k * 1024 + tid];
        sz[tid] = acc;
        __syncthreads();
        if (tid < EU) {
            float ig = sigf(sz[tid]);
            float fg = sigf(sz[EU + tid]);
            float gg = tanhf(sz[2 * EU + tid]);
            float og = sigf(sz[3 * EU + tid]);
            float cc = fg * sc[tid] + ig * gg;
            sc[tid] = cc;
            sh[tid] = og * tanhf(cc);
        }
        __syncthreads();
    }
    if (tid < EU) { hT[tid] = sh[tid]; cT[tid] = sc[tid]; }
}

// ---------------- state reducers: h = [sh,wh]@rh_w + rh_b ; c likewise ----------------
__global__ __launch_bounds__(256) void k_reduce_states(
    const float* __restrict__ shv, const float* __restrict__ whv,
    const float* __restrict__ scv, const float* __restrict__ wcv,
    const float* __restrict__ rhw, const float* __restrict__ rhb,
    const float* __restrict__ rcw, const float* __restrict__ rcb,
    float* __restrict__ h, float* __restrict__ c) {
    int j = threadIdx.x;
    float ah = rhb[j], ac = rcb[j];
    for (int k = 0; k < EU; ++k) {
        ah += shv[k] * rhw[k * EU + j];
        ac += scv[k] * rcw[k * EU + j];
    }
    for (int k = 0; k < EU; ++k) {
        ah += whv[k] * rhw[(EU + k) * EU + j];
        ac += wcv[k] * rcw[(EU + k) * EU + j];
    }
    h[j] = ah;
    c[j] = ac;
}

// ---------------- persistent decoder: 299 outer x 128 inner LSTM steps, WMMA proj + softmax ----------------
__global__ __launch_bounds__(1024) void k_decoder(
    const float* __restrict__ emb,   // [128, 256]
    const float* __restrict__ dWi,   // [256, 1024]
    const float* __restrict__ dWh,   // [256, 1024]
    const float* __restrict__ db,    // [1024]
    const h16*  __restrict__ outwT,  // [128, 256] f16 (transposed out_w)
    const float* __restrict__ outb,  // [128]
    const float* __restrict__ dec,   // [300, 128] one-hot floats
    const float* __restrict__ h0, const float* __restrict__ c0,
    h16* __restrict__ hs,            // ws [128, 256] f16
    float* __restrict__ proj,        // ws [128, 128] f32
    float* __restrict__ out) {       // [128, 300, 128]
    __shared__ float sh[DU], sc[DU], sz[1024], szsel[2][1024];
    int tid = threadIdx.x;
    int wave = tid >> 5, lane = tid & 31;
    // inner input is emb[0] or emb[1] (int-cast one-hot) -> precompute both x@Wi+b rows
#pragma unroll
    for (int r = 0; r < 2; ++r) {
        float acc = db[tid];
        for (int k = 0; k < DU; ++k) acc += emb[r * DU + k] * dWi[k * 1024 + tid];
        szsel[r][tid] = acc;
    }
    if (tid < DU) { sh[tid] = h0[tid]; sc[tid] = c0[tid]; }
    __syncthreads();
    int lr = lane & 15;
    int kh = (lane < 16) ? 0 : 16;
    int mb = (lane < 16) ? 0 : 8;
    for (int t = 0; t < LL - 1; ++t) {
        for (int v = 0; v < VV; ++v) {
            int sel = ((int)dec[t * VV + v]) & 1;
            float acc = szsel[sel][tid];
#pragma unroll 4
            for (int k = 0; k < DU; ++k) acc += sh[k] * dWh[k * 1024 + tid];
            sz[tid] = acc;
            __syncthreads();
            if (tid < DU) {
                float ig = sigf(sz[tid]);
                float fg = sigf(sz[DU + tid]);
                float gg = tanhf(sz[2 * DU + tid]);
                float og = sigf(sz[3 * DU + tid]);
                float cc = fg * sc[tid] + ig * gg;
                float hh = og * tanhf(cc);
                sc[tid] = cc;
                sh[tid] = hh;
                hs[v * DU + tid] = (h16)hh;
            }
            __syncthreads();
        }
        __threadfence();
        __syncthreads();
        // projection: proj[128,128] = hs[128,256] @ out_w + outb  (WMMA, 64 tiles / 32 waves)
        for (int tt = wave; tt < 64; tt += 32) {
            int tm = (tt >> 3) * 16, tn = (tt & 7) * 16;
            v8f acc = {};
            for (int kk = 0; kk < DU; kk += 32) {
                v16h a, b;
#pragma unroll
                for (int i = 0; i < 16; ++i) a[i] = hs[(tm + lr) * DU + kk + kh + i];
#pragma unroll
                for (int i = 0; i < 16; ++i) b[i] = outwT[(tn + lr) * DU + kk + kh + i];
                acc = __builtin_amdgcn_wmma_f32_16x16x32_f16(false, a, false, b, (short)0, acc, false, false);
            }
#pragma unroll
            for (int r = 0; r < 8; ++r)
                proj[(tm + mb + r) * VV + tn + lr] = acc[r] + outb[tn + lr];
        }
        __threadfence();
        __syncthreads();
        // softmax over vocab + transposed store: out[v, t, :]
        if (tid < VV) {
            const float* row = proj + tid * VV;
            float mx = row[0];
            for (int u = 1; u < VV; ++u) mx = fmaxf(mx, row[u]);
            float s = 0.f;
            for (int u = 0; u < VV; ++u) s += __expf(row[u] - mx);
            float inv = 1.f / s;
            float* dst = out + (size_t)tid * LL * VV + (size_t)t * VV;
            for (int u = 0; u < VV; ++u) dst[u] = __expf(row[u] - mx) * inv;
        }
        __syncthreads();
    }
}

// =====================================================================================
extern "C" void kernel_launch(void* const* d_in, const int* in_sizes, int n_in,
                              void* d_out, int out_size, void* d_ws, size_t ws_size,
                              hipStream_t stream) {
    const float* stft   = (const float*)d_in[0];
    const float* wavef  = (const float*)d_in[1];
    const float* dec    = (const float*)d_in[2];
    const float* s_cw1  = (const float*)d_in[3];
    const float* s_cb1  = (const float*)d_in[4];
    const float* s_cw2  = (const float*)d_in[5];
    const float* s_cb2  = (const float*)d_in[6];
    const float* s_Wi   = (const float*)d_in[7];
    const float* s_Wh   = (const float*)d_in[8];
    const float* s_b    = (const float*)d_in[9];
    const float* w_cw1  = (const float*)d_in[10];
    const float* w_cb1  = (const float*)d_in[11];
    const float* w_cw2  = (const float*)d_in[12];
    const float* w_cb2  = (const float*)d_in[13];
    const float* w_Wi   = (const float*)d_in[14];
    const float* w_Wh   = (const float*)d_in[15];
    const float* w_b    = (const float*)d_in[16];
    const float* rh_w   = (const float*)d_in[17];
    const float* rh_b   = (const float*)d_in[18];
    const float* rc_w   = (const float*)d_in[19];
    const float* rc_b   = (const float*)d_in[20];
    const float* emb    = (const float*)d_in[21];
    const float* d_Wi   = (const float*)d_in[22];
    const float* d_Wh   = (const float*)d_in[23];
    const float* d_b    = (const float*)d_in[24];
    const float* out_w  = (const float*)d_in[25];
    const float* out_b  = (const float*)d_in[26];
    float* out = (float*)d_out;

    // ---- workspace arena ----
    char* base = (char*)d_ws;
    size_t off = 0;
    auto alloc = [&](size_t bytes) -> char* {
        char* p = base + off;
        off += (bytes + 255) & ~size_t(255);
        return p;
    };
    float* ws_c1   = (float*)alloc((size_t)S_C1H * S_C1W * CC * 4);
    h16*   ws_p1h  = (h16*)  alloc((size_t)S_P1H * S_P1W * CC * 2);
    h16*   ws_w2T  = (h16*)  alloc((size_t)9 * CC * CC * 2);        // [256][2304]
    h16*   ws_A2   = (h16*)  alloc((size_t)S_C2M * 9 * CC * 2);
    float* ws_c2   = (float*)alloc((size_t)S_C2M * CC * 4);
    float* ws_p2   = (float*)alloc((size_t)TS * FLATD * 4);
    float* ws_zxs  = (float*)alloc((size_t)TS * 1024 * 4);
    float* ws_sh   = (float*)alloc(256 * 4);
    float* ws_sc   = (float*)alloc(256 * 4);
    h16*   ws_Aw1  = (h16*)  alloc((size_t)W_C1 * 3072 * 2);
    h16*   ws_w1T  = (h16*)  alloc((size_t)3 * 1024 * 256 * 2);     // [256][3072]
    float* ws_wc1  = (float*)alloc((size_t)W_C1 * 256 * 4);
    h16*   ws_wp1h = (h16*)  alloc((size_t)W_P1 * 256 * 2);
    h16*   ws_Aw2  = (h16*)  alloc((size_t)W_C2 * 768 * 2);
    h16*   ws_w3T  = (h16*)  alloc((size_t)3 * 256 * 256 * 2);      // [256][768]
    float* ws_wc2  = (float*)alloc((size_t)W_C2 * 256 * 4);
    h16*   ws_wp2h = (h16*)  alloc((size_t)W_P2 * 256 * 2);
    h16*   ws_wWiT = (h16*)  alloc((size_t)256 * 1024 * 2);         // [1024][256]
    float* ws_zxw  = (float*)alloc((size_t)W_P2 * 1024 * 4);
    float* ws_wh   = (float*)alloc(256 * 4);
    float* ws_wc   = (float*)alloc(256 * 4);
    float* ws_h0   = (float*)alloc(256 * 4);
    float* ws_c0   = (float*)alloc(256 * 4);
    h16*   ws_owT  = (h16*)  alloc((size_t)128 * 256 * 2);          // [128][256]
    h16*   ws_hs   = (h16*)  alloc((size_t)128 * 256 * 2);
    float* ws_proj = (float*)alloc((size_t)128 * 128 * 4);
    (void)ws_size; (void)n_in; (void)in_sizes;

    auto cdiv = [](long a, long b) { return (int)((a + b - 1) / b); };

    // ---- output: zero (final L slot stays zero) ----
    k_zero_f32<<<cdiv(out_size, 256), 256, 0, stream>>>(out, out_size);

    // ---- STFT encoder ----
    k_conv2d_s1<<<cdiv((long)S_C1H * S_C1W * CC, 256), 256, 0, stream>>>(stft, s_cw1, s_cb1, ws_c1);
    k_maxpool2d<<<cdiv((long)S_P1H * S_P1W * CC, 256), 256, 0, stream>>>(ws_c1, nullptr, ws_p1h, S_C1H, S_C1W, CC);
    k_transpose_f32_to_f16<<<cdiv(9 * CC * CC, 256), 256, 0, stream>>>(s_cw2, ws_w2T, 9 * CC, CC);
    k_im2col2d_f16<<<cdiv((long)S_C2M * 9 * CC, 256), 256, 0, stream>>>(ws_p1h, ws_A2, S_P1W, CC, S_C2H, S_C2W);
    {
        dim3 g(cdiv(CC, 128), cdiv(S_C2M, 32));
        k_gemm_f16_wmma<<<g, 256, 0, stream>>>(ws_A2, ws_w2T, s_cb2, ws_c2, S_C2M, CC, 9 * CC, 1);
    }
    k_maxpool2d<<<cdiv((long)S_P2H * S_P2W * CC, 256), 256, 0, stream>>>(ws_c2, ws_p2, nullptr, S_C2H, S_C2W, CC);
    k_stft_zpre<<<4, 256, 0, stream>>>(ws_p2, s_Wi, s_b, ws_zxs);
    k_lstm_seq<<<1, 1024, 0, stream>>>(ws_zxs, s_Wh, ws_sh, ws_sc, TS);

    // ---- Waveform encoder ----
    k_im2col1d_f32<<<cdiv((long)W_C1 * 3072, 256), 256, 0, stream>>>(wavef, ws_Aw1, 1024, W_C1);
    k_transpose_f32_to_f16<<<cdiv(3 * 1024 * 256, 256), 256, 0, stream>>>(w_cw1, ws_w1T, 3072, 256);
    {
        dim3 g(cdiv(256, 128), cdiv(W_C1, 32));
        k_gemm_f16_wmma<<<g, 256, 0, stream>>>(ws_Aw1, ws_w1T, w_cb1, ws_wc1, W_C1, 256, 3072, 1);
    }
    k_maxpool1d<<<cdiv((long)W_P1 * 256, 256), 256, 0, stream>>>(ws_wc1, nullptr, ws_wp1h, W_C1, 256);
    k_im2col1d_f16<<<cdiv((long)W_C2 * 768, 256), 256, 0, stream>>>(ws_wp1h, ws_Aw2, 256, W_C2);
    k_transpose_f32_to_f16<<<cdiv(3 * 256 * 256, 256), 256, 0, stream>>>(w_cw2, ws_w3T, 768, 256);
    {
        dim3 g(cdiv(256, 128), cdiv(W_C2, 32));
        k_gemm_f16_wmma<<<g, 256, 0, stream>>>(ws_Aw2, ws_w3T, w_cb2, ws_wc2, W_C2, 256, 768, 1);
    }
    k_maxpool1d<<<cdiv((long)W_P2 * 256, 256), 256, 0, stream>>>(ws_wc2, nullptr, ws_wp2h, W_C2, 256);
    k_transpose_f32_to_f16<<<cdiv(256 * 1024, 256), 256, 0, stream>>>(w_Wi, ws_wWiT, 256, 1024);
    {
        dim3 g(cdiv(1024, 128), cdiv(W_P2, 32));
        k_gemm_f16_wmma<<<g, 256, 0, stream>>>(ws_wp2h, ws_wWiT, w_b, ws_zxw, W_P2, 1024, 256, 0);
    }
    k_lstm_seq<<<1, 1024, 0, stream>>>(ws_zxw, w_Wh, ws_wh, ws_wc, W_P2);

    // ---- state reducers ----
    k_reduce_states<<<1, 256, 0, stream>>>(ws_sh, ws_wh, ws_sc, ws_wc,
                                           rh_w, rh_b, rc_w, rc_b, ws_h0, ws_c0);

    // ---- decoder (persistent, sequential chain) ----
    k_transpose_f32_to_f16<<<cdiv(256 * 128, 256), 256, 0, stream>>>(out_w, ws_owT, 256, 128);
    k_decoder<<<1, 1024, 0, stream>>>(emb, d_Wi, d_Wh, d_b, ws_owT, out_b, dec,
                                      ws_h0, ws_c0, ws_hs, ws_proj, out);
}
